// G_CAM_Module_49520972922893
// MI455X (gfx1250) — matrix operand order repaired and checked by reading, not compiled, not run on hardware
//
#include <hip/hip_runtime.h>
#include <math.h>

#define BATCH 16
#define CCH   64
#define HDIM  256
#define NPIX  (HDIM * HDIM)      // 65536

// ---- CDNA5 WMMA types ----
typedef _Float16 v8h  __attribute__((ext_vector_type(8)));
typedef _Float16 v16h __attribute__((ext_vector_type(16)));
typedef float    v8f  __attribute__((ext_vector_type(8)));

static __device__ __forceinline__ v8f wmma16(v16h a, v16h b, v8f c) {
  // (neg_a, A, neg_b, B, c_mod, C, reuse_a, reuse_b)
  return __builtin_amdgcn_wmma_f32_16x16x32_f16(false, a, false, b, (short)0, c,
                                                false, false);
}

#define HSTRIDE 72   // halves per row: 144B, 16B-aligned rows for b128 LDS loads

// A fragment (16x32, M=row across lanes): elems 0-7 = K 8h+0..7, 8-15 = K 16+8h+0..7
static __device__ __forceinline__ v16h fragA(const _Float16* row, int kb, int half) {
  v8h p0 = *reinterpret_cast<const v8h*>(row + kb + 8 * half);
  v8h p1 = *reinterpret_cast<const v8h*>(row + kb + 16 + 8 * half);
  return __builtin_shufflevector(p0, p1, 0, 1, 2, 3, 4, 5, 6, 7, 8, 9, 10, 11,
                                 12, 13, 14, 15);
}
// B fragment (32x16, N=row across lanes, operand staged K-contiguous):
// elems 0-7 = K 8h+0..7, 8-15 = K 16+8h+0..7
static __device__ __forceinline__ v16h fragB(const _Float16* row, int kb, int half) {
  v8h p0 = *reinterpret_cast<const v8h*>(row + kb + 8 * half);
  v8h p1 = *reinterpret_cast<const v8h*>(row + kb + 16 + 8 * half);
  return __builtin_shufflevector(p0, p1, 0, 1, 2, 3, 4, 5, 6, 7, 8, 9, 10, 11,
                                 12, 13, 14, 15);
}

static __device__ __forceinline__ void split_f32(float x, _Float16& hi, _Float16& lo) {
  hi = (_Float16)x;
  lo = (_Float16)(x - (float)hi);
}

// =====================================================================
// Kernel 1: partial Gram  E[b] += Xslice * Xslice^T  (split-f16 WMMA,
// atomic reduce).  grid = (GR_SLICES, BATCH), block = 256 (8 waves)
// =====================================================================
#define GR_SLICES 64
#define GR_NCOLS  (NPIX / GR_SLICES)   // 1024 columns per WG
#define GR_CHUNK  64

__global__ __launch_bounds__(256) void gram_kernel(const float* __restrict__ X,
                                                   float* __restrict__ E) {
  __shared__ __align__(16) _Float16 Xhi[CCH][HSTRIDE];
  __shared__ __align__(16) _Float16 Xlo[CCH][HSTRIDE];

  const int b     = blockIdx.y;
  const int slice = blockIdx.x;
  const int tid   = threadIdx.x;
  const int lane  = tid & 31;
  const int wave  = tid >> 5;
  const int half  = lane >> 4;
  const int m     = lane & 15;

  // wave w owns tiles t=2w,2w+1 of the 4x4 output tile grid (shared ci)
  const int t0  = wave * 2;
  const int ci  = t0 >> 2;
  const int di0 = t0 & 3;
  const int di1 = di0 + 1;

  const float* Xb = X + (size_t)b * CCH * NPIX + (size_t)slice * GR_NCOLS;

  v8f acc0 = {0.f, 0.f, 0.f, 0.f, 0.f, 0.f, 0.f, 0.f};
  v8f acc1 = {0.f, 0.f, 0.f, 0.f, 0.f, 0.f, 0.f, 0.f};

  for (int chunk = 0; chunk < GR_NCOLS; chunk += GR_CHUNK) {
    // stage 64x64 fp32 -> hi/lo f16 in LDS
    for (int f = tid; f < CCH * (GR_CHUNK / 4); f += 256) {
      const int row = f >> 4;
      const int c4  = (f & 15) << 2;
      const float4 v =
          *reinterpret_cast<const float4*>(Xb + (size_t)row * NPIX + chunk + c4);
      const float t[4] = {v.x, v.y, v.z, v.w};
#pragma unroll
      for (int i = 0; i < 4; ++i) {
        _Float16 h, l;
        split_f32(t[i], h, l);
        Xhi[row][c4 + i] = h;
        Xlo[row][c4 + i] = l;
      }
    }
    __syncthreads();

#pragma unroll
    for (int kb = 0; kb < GR_CHUNK; kb += 32) {
      // Gram: B-fragment of channel-block d is a row run of X, like A
      v16h Ah = fragA(&Xhi[ci * 16 + m][0], kb, half);
      v16h Al = fragA(&Xlo[ci * 16 + m][0], kb, half);
      v16h B0h = fragB(&Xhi[di0 * 16 + m][0], kb, half);
      v16h B0l = fragB(&Xlo[di0 * 16 + m][0], kb, half);
      v16h B1h = fragB(&Xhi[di1 * 16 + m][0], kb, half);
      v16h B1l = fragB(&Xlo[di1 * 16 + m][0], kb, half);
      acc0 = wmma16(Ah, B0h, acc0);   // hi*hi
      acc0 = wmma16(Ah, B0l, acc0);   // hi*lo
      acc0 = wmma16(Al, B0h, acc0);   // lo*hi
      acc1 = wmma16(Ah, B1h, acc1);
      acc1 = wmma16(Ah, B1l, acc1);
      acc1 = wmma16(Al, B1h, acc1);
    }
    __syncthreads();
  }

  // C/D layout: VGPR v, lanes<16 -> M=v, lanes>=16 -> M=v+8, N=lane&15
  float* Eb = E + (size_t)b * CCH * CCH;
#pragma unroll
  for (int v = 0; v < 8; ++v) {
    const int row = ci * 16 + v + half * 8;
    atomicAdd(&Eb[row * CCH + di0 * 16 + m], acc0[v]);
    atomicAdd(&Eb[row * CCH + di1 * 16 + m], acc1[v]);
  }
}

// =====================================================================
// Kernel 2: per-batch attention chain (64x64)
//   Ax = softmax(Ex); Ag = softmax(Eg) stored transposed;
//   GE = Ax*Ag (split-f16 WMMA); GA = softmax(rowmax(GE)-GE)
// grid = BATCH, block = 256 (8 waves)
// =====================================================================
#define GESTRIDE 68

__global__ __launch_bounds__(256) void attn_kernel(const float* __restrict__ Ex,
                                                   const float* __restrict__ Eg,
                                                   float* __restrict__ GA) {
  __shared__ __align__(16) _Float16 Axh[CCH][HSTRIDE];
  __shared__ __align__(16) _Float16 Axl[CCH][HSTRIDE];
  __shared__ __align__(16) _Float16 AgTh[CCH][HSTRIDE];  // transposed guide attn
  __shared__ __align__(16) _Float16 AgTl[CCH][HSTRIDE];
  __shared__ float GE[CCH][GESTRIDE];

  const int b   = blockIdx.x;
  const int tid = threadIdx.x;

  // ---- phase 1: row softmax of both energies -> hi/lo f16 LDS ----
  if (tid < 128) {
    const int r = tid & 63;
    const float* src =
        (tid < 64 ? Ex : Eg) + (size_t)b * CCH * CCH + (size_t)r * CCH;
    float mx = -__builtin_inff();
    for (int j = 0; j < CCH; ++j) mx = fmaxf(mx, src[j]);
    float s = 0.f;
    for (int j = 0; j < CCH; ++j) s += __expf(src[j] - mx);
    const float inv = 1.f / s;
    for (int j = 0; j < CCH; ++j) {
      const float p = __expf(src[j] - mx) * inv;
      _Float16 h, l;
      split_f32(p, h, l);
      if (tid < 64) { Axh[r][j] = h;  Axl[r][j] = l; }
      else          { AgTh[j][r] = h; AgTl[j][r] = l; }  // store transposed
    }
  }
  __syncthreads();

  // ---- phase 2: GE = Ax * Ag via split-f16 WMMA (16 tiles / 8 waves) ----
  const int lane = tid & 31;
  const int wave = tid >> 5;
  const int half = lane >> 4;
  const int m    = lane & 15;
  const int t0   = wave * 2;
  const int ci   = t0 >> 2;
  const int di0  = t0 & 3;
  const int di1  = di0 + 1;

  v8f acc0 = {0.f, 0.f, 0.f, 0.f, 0.f, 0.f, 0.f, 0.f};
  v8f acc1 = {0.f, 0.f, 0.f, 0.f, 0.f, 0.f, 0.f, 0.f};

#pragma unroll
  for (int kb = 0; kb < CCH; kb += 32) {
    v16h Ah  = fragA(&Axh[ci * 16 + m][0], kb, half);
    v16h Al  = fragA(&Axl[ci * 16 + m][0], kb, half);
    v16h B0h = fragB(&AgTh[di0 * 16 + m][0], kb, half);
    v16h B0l = fragB(&AgTl[di0 * 16 + m][0], kb, half);
    v16h B1h = fragB(&AgTh[di1 * 16 + m][0], kb, half);
    v16h B1l = fragB(&AgTl[di1 * 16 + m][0], kb, half);
    acc0 = wmma16(Ah, B0h, acc0);
    acc0 = wmma16(Ah, B0l, acc0);
    acc0 = wmma16(Al, B0h, acc0);
    acc1 = wmma16(Ah, B1h, acc1);
    acc1 = wmma16(Ah, B1l, acc1);
    acc1 = wmma16(Al, B1h, acc1);
  }
#pragma unroll
  for (int v = 0; v < 8; ++v) {
    const int row = ci * 16 + v + half * 8;
    GE[row][di0 * 16 + m] = acc0[v];
    GE[row][di1 * 16 + m] = acc1[v];
  }
  __syncthreads();

  // ---- phase 3: GA = softmax(rowmax - GE) -> workspace ----
  if (tid < 64) {
    const int r = tid;
    float mx = -__builtin_inff();
    for (int j = 0; j < CCH; ++j) mx = fmaxf(mx, GE[r][j]);
    float nmax = -__builtin_inff();
    for (int j = 0; j < CCH; ++j) nmax = fmaxf(nmax, mx - GE[r][j]);
    float s = 0.f;
    for (int j = 0; j < CCH; ++j) {
      const float e = __expf((mx - GE[r][j]) - nmax);
      GE[r][j] = e;
      s += e;
    }
    const float inv = 1.f / s;
    float* dst = GA + (size_t)b * CCH * CCH + (size_t)r * CCH;
    for (int j = 0; j < CCH; ++j) dst[j] = GE[r][j] * inv;
  }
}

// =====================================================================
// Kernel 3: out = gamma * (GA * X) + x   (split-f16 WMMA, fp32 residual)
// grid = (NPIX/AP_NTILE, BATCH), block = 256 (8 waves)
// =====================================================================
#define AP_NTILE 128

__global__ __launch_bounds__(256) void apply_kernel(
    const float* __restrict__ X, const float* __restrict__ GA,
    const float* __restrict__ gamma, float* __restrict__ Out) {
  __shared__ __align__(16) _Float16 GAh[CCH][HSTRIDE];
  __shared__ __align__(16) _Float16 GAl[CCH][HSTRIDE];
  __shared__ __align__(16) _Float16 XTh[AP_NTILE][HSTRIDE];  // X tile, transposed
  __shared__ __align__(16) _Float16 XTl[AP_NTILE][HSTRIDE];

  const int b     = blockIdx.y;
  const int nbase = blockIdx.x * AP_NTILE;
  const int tid   = threadIdx.x;

  // stage GA (64x64) rows -> hi/lo
  const float* GAb = GA + (size_t)b * CCH * CCH;
  for (int f = tid; f < CCH * (CCH / 4); f += 256) {
    const int row = f >> 4;
    const int c4  = (f & 15) << 2;
    const float4 v = *reinterpret_cast<const float4*>(GAb + row * CCH + c4);
    const float t[4] = {v.x, v.y, v.z, v.w};
#pragma unroll
    for (int i = 0; i < 4; ++i) {
      _Float16 h, l;
      split_f32(t[i], h, l);
      GAh[row][c4 + i] = h;
      GAl[row][c4 + i] = l;
    }
  }
  // stage X tile (64 ch x 128 cols) transposed -> hi/lo (B operand K-contig)
  const float* Xb = X + (size_t)b * CCH * NPIX + nbase;
  for (int f = tid; f < CCH * (AP_NTILE / 4); f += 256) {
    const int row = f >> 5;            // channel
    const int c4  = (f & 31) << 2;     // n offset
    const float4 v =
        *reinterpret_cast<const float4*>(Xb + (size_t)row * NPIX + c4);
    const float t[4] = {v.x, v.y, v.z, v.w};
#pragma unroll
    for (int i = 0; i < 4; ++i) {
      _Float16 h, l;
      split_f32(t[i], h, l);
      XTh[c4 + i][row] = h;
      XTl[c4 + i][row] = l;
    }
  }
  __syncthreads();

  const int lane = tid & 31;
  const int wave = tid >> 5;
  const int half = lane >> 4;
  const int m    = lane & 15;
  const int ci   = wave >> 1;        // channel block 0..3
  const int ns0  = (wave & 1) * 4;   // 4 of 8 column sub-tiles

  v8f acc[4];
#pragma unroll
  for (int s = 0; s < 4; ++s)
    acc[s] = (v8f){0.f, 0.f, 0.f, 0.f, 0.f, 0.f, 0.f, 0.f};

#pragma unroll
  for (int kb = 0; kb < CCH; kb += 32) {
    v16h Ah = fragA(&GAh[ci * 16 + m][0], kb, half);
    v16h Al = fragA(&GAl[ci * 16 + m][0], kb, half);
#pragma unroll
    for (int s = 0; s < 4; ++s) {
      const int c0 = (ns0 + s) * 16;
      v16h Bh = fragB(&XTh[c0 + m][0], kb, half);
      v16h Bl = fragB(&XTl[c0 + m][0], kb, half);
      acc[s] = wmma16(Ah, Bh, acc[s]);
      acc[s] = wmma16(Ah, Bl, acc[s]);
      acc[s] = wmma16(Al, Bh, acc[s]);
    }
  }

  const float gm = gamma[0];
  float* Ob = Out + (size_t)b * CCH * NPIX + nbase;
#pragma unroll
  for (int s = 0; s < 4; ++s) {
    const int lc = (ns0 + s) * 16 + m;
#pragma unroll
    for (int v = 0; v < 8; ++v) {
      const int c = ci * 16 + v + half * 8;
      // fp32 residual straight from global (L2-hot: tile just streamed)
      Ob[(size_t)c * NPIX + lc] = gm * acc[s][v] + Xb[(size_t)c * NPIX + lc];
    }
  }
}

// =====================================================================
extern "C" void kernel_launch(void* const* d_in, const int* in_sizes, int n_in,
                              void* d_out, int out_size, void* d_ws,
                              size_t ws_size, hipStream_t stream) {
  const float* x     = (const float*)d_in[0];
  const float* g     = (const float*)d_in[1];
  const float* gamma = (const float*)d_in[2];
  float* out = (float*)d_out;

  float* ws = (float*)d_ws;
  float* Ex = ws;                                 // B*64*64
  float* Eg = ws + (size_t)BATCH * CCH * CCH;     // B*64*64
  float* GA = ws + (size_t)2 * BATCH * CCH * CCH;

  // zero the atomic-accumulated energy buffers (graph-capture safe)
  (void)hipMemsetAsync(ws, 0, (size_t)2 * BATCH * CCH * CCH * sizeof(float),
                       stream);

  dim3 gg(GR_SLICES, BATCH);
  gram_kernel<<<gg, 256, 0, stream>>>(x, Ex);
  gram_kernel<<<gg, 256, 0, stream>>>(g, Eg);
  attn_kernel<<<BATCH, 256, 0, stream>>>(Ex, Eg, GA);
  apply_kernel<<<dim3(NPIX / AP_NTILE, BATCH), 256, 0, stream>>>(x, GA, gamma,
                                                                 out);
}
